// Attention_53257594471037
// MI455X (gfx1250) — matrix operand order repaired and checked by reading, not compiled, not run on hardware
//
#include <hip/hip_runtime.h>
#include <hip/hip_bf16.h>

// Problem constants (match reference)
#define BATCH    8
#define SEQ      1024
#define EMBED    512
#define CLS      512
#define XW       1024            // EMBED + CLS
#define HEADS    8
#define DIM_HEAD 64
#define INNER    512
#define V_DIM    1024
#define DV       128
#define ROWS     8192            // BATCH*SEQ
#define SCALE    0.125f          // 64^-0.5

typedef __attribute__((ext_vector_type(16))) __bf16 bf16x16;
typedef __attribute__((ext_vector_type(8)))  __bf16 bf16x8;
typedef __attribute__((ext_vector_type(8)))  float  f32x8;
typedef __attribute__((ext_vector_type(4)))  float  f32x4;

union Frag16 { bf16x16 v; bf16x8 h[2]; };

#if defined(__AMDGCN__) && __has_builtin(__builtin_amdgcn_tensor_load_to_lds) && \
    __has_builtin(__builtin_amdgcn_s_wait_tensorcnt)
#define HAVE_TDM 1
#else
#define HAVE_TDM 0
#endif

__device__ __forceinline__ __bf16 f2bf(float f) {
  unsigned u = __builtin_bit_cast(unsigned, f);
  unsigned r = u + 0x7FFFu + ((u >> 16) & 1u);      // round-to-nearest-even
  unsigned short s = (unsigned short)(r >> 16);
  return __builtin_bit_cast(__bf16, s);
}

__device__ __forceinline__ f32x8 zero8() {
  f32x8 z;
  for (int i = 0; i < 8; ++i) z[i] = 0.0f;
  return z;
}

__device__ __forceinline__ f32x8 wmma_bf16(const Frag16& a, const Frag16& b, f32x8 c) {
  return __builtin_amdgcn_wmma_f32_16x16x32_bf16(
      false, a.v, false, b.v, (short)0, c, false, false);
}

#if HAVE_TDM
typedef __attribute__((ext_vector_type(4))) unsigned u32x4;
typedef __attribute__((ext_vector_type(4))) int      i32x4;
typedef __attribute__((ext_vector_type(8))) int      i32x8;

// TDM: DMA one 128-row x 32-elem bf16 tile, global(row stride = stride_elems)
// -> LDS rows padded to 40 elems (pad 4 DWORDs after every 16 DWORDs).
__device__ __forceinline__ void tdm_load_tile(const __bf16* gsrc, unsigned lds_off,
                                              unsigned stride_elems) {
  unsigned long long ga = (unsigned long long)(size_t)gsrc;
  u32x4 g0;
  g0[0] = 1u;                                                  // count=1 (user D#)
  g0[1] = lds_off;                                             // LDS byte address
  g0[2] = (unsigned)ga;                                        // global_addr[31:0]
  g0[3] = ((unsigned)(ga >> 32) & 0x01FFFFFFu) | 0x80000000u;  // ga[56:32], type=2
  i32x8 g1;
  // data_size=1(2B) | pad_enable | pad_interval=3(16 dw) | pad_amount=3(4 dw)
  g1[0] = (int)((1u << 16) | (1u << 20) | (3u << 22) | (3u << 25));
  g1[1] = 0;                                // abar=0; tensor_dim0[15:0]=0 (td0=1M)
  g1[2] = 0x10;                             // td0[31:16]=0x10; td1[15:0]=0
  g1[3] = (int)(0x10u | (32u << 16));       // td1[31:16]=0x10; tile_dim0=32
  g1[4] = 128;                              // tile_dim1=128; tile_dim2=0
  g1[5] = (int)stride_elems;                // tensor_dim0_stride[31:0]
  g1[6] = 0;                                // stride hi / dim1_stride lo
  g1[7] = 0;
  i32x4 gz; gz[0] = 0; gz[1] = 0; gz[2] = 0; gz[3] = 0;
  i32x8 gz8;
  for (int i = 0; i < 8; ++i) gz8[i] = 0;
  // 6-arg form (toolchain-confirmed by round-2 diagnostic)
  __builtin_amdgcn_tensor_load_to_lds(g0, g1, gz, gz, gz8, 0);
}
#endif

// ---------------------------------------------------------------------------
// 1) fp32 -> bf16 elementwise convert (x)
// ---------------------------------------------------------------------------
__global__ __launch_bounds__(256) void k_cvt_bf16(const float* __restrict__ in,
                                                  __bf16* __restrict__ out) {
  size_t i = ((size_t)blockIdx.x * blockDim.x + threadIdx.x) * 8;
  f32x4 a = *(const f32x4*)(in + i);
  f32x4 b = *(const f32x4*)(in + i + 4);
  bf16x8 o;
  o[0] = f2bf(a[0]); o[1] = f2bf(a[1]); o[2] = f2bf(a[2]); o[3] = f2bf(a[3]);
  o[4] = f2bf(b[0]); o[5] = f2bf(b[1]); o[6] = f2bf(b[2]); o[7] = f2bf(b[3]);
  *(bf16x8*)(out + i) = o;
}

// ---------------------------------------------------------------------------
// 2) fp32 [K][N] -> bf16 [N][K] tiled transpose (weights)
// ---------------------------------------------------------------------------
__global__ __launch_bounds__(256) void k_transpose_bf16(const float* __restrict__ in,
                                                        __bf16* __restrict__ out,
                                                        int K, int N) {
  __shared__ __bf16 tile[32][33];
  int nb = blockIdx.x * 32, kb = blockIdx.y * 32;
  int tx = threadIdx.x & 31, ty = threadIdx.x >> 5;   // 8 row-groups
  for (int i = ty; i < 32; i += 8)
    tile[i][tx] = f2bf(in[(size_t)(kb + i) * N + nb + tx]);
  __syncthreads();
  for (int i = ty; i < 32; i += 8)
    out[(size_t)(nb + i) * K + kb + tx] = tile[tx][i];
}

// ---------------------------------------------------------------------------
// 3) bf16 GEMM: C[M x 1024] = A[M x K](lda) * Bt[1024 x K]^T + bias
//    WG: 256 thr (8 waves), 128x128 tile; wave: 64x32 = 8 wmma accums.
//    TDM path: double-buffered LDS, wave0 issues tensor_load_to_lds for the
//    next tile while all waves run WMMAs on the current one (1 barrier/step).
// ---------------------------------------------------------------------------
__global__ __launch_bounds__(256) void k_gemm_bf16(const __bf16* __restrict__ A, int lda,
                                                   const __bf16* __restrict__ Bt,
                                                   const float* __restrict__ bias,
                                                   void* __restrict__ Cout, int outF32,
                                                   int K) {
  __shared__ __attribute__((aligned(16))) __bf16 As[2][128 * 40];
  __shared__ __attribute__((aligned(16))) __bf16 Bs[2][128 * 40];

  const int tid  = threadIdx.x;
  const int lane = tid & 31, wave = tid >> 5;
  const int mwg = blockIdx.y * 128, nwg = blockIdx.x * 128;
  const int wm = (wave >> 2) * 64;   // wave M offset: 0 / 64
  const int wn = (wave & 3) * 32;    // wave N offset: 0..96

  const int m15  = lane & 15;
  const int khi  = lane >> 4;        // 0/1
  const int klo8 = khi * 8;

  f32x8 acc[4][2];
  for (int mt = 0; mt < 4; ++mt)
    for (int nt = 0; nt < 2; ++nt) acc[mt][nt] = zero8();

#if HAVE_TDM
  const int T = K >> 5;
  const __bf16* Ab = A  + (size_t)mwg * lda;
  const __bf16* Bb = Bt + (size_t)nwg * K;
  if (wave == 0) {
    tdm_load_tile(Ab, (unsigned)(size_t)(const void*)As[0], (unsigned)lda);
    tdm_load_tile(Bb, (unsigned)(size_t)(const void*)Bs[0], (unsigned)K);
  }
  for (int t = 0; t < T; ++t) {
    const int cur = t & 1;
    if (wave == 0) __builtin_amdgcn_s_wait_tensorcnt(0);  // tile t landed in LDS
    __syncthreads();                                      // ...and visible to all
    if (wave == 0 && t + 1 < T) {                         // prefetch tile t+1
      tdm_load_tile(Ab + (t + 1) * 32, (unsigned)(size_t)(const void*)As[cur ^ 1],
                    (unsigned)lda);
      tdm_load_tile(Bb + (t + 1) * 32, (unsigned)(size_t)(const void*)Bs[cur ^ 1],
                    (unsigned)K);
    }
    const __bf16* Asb = As[cur];
    const __bf16* Bsb = Bs[cur];
    Frag16 aF[4], bF[2];
    for (int mt = 0; mt < 4; ++mt) {
      const __bf16* p = Asb + (wm + mt * 16 + m15) * 40;
      aF[mt].h[0] = *(const bf16x8*)(p + khi * 16);
      aF[mt].h[1] = *(const bf16x8*)(p + khi * 16 + 8);
    }
    for (int nt = 0; nt < 2; ++nt) {
      const __bf16* p = Bsb + (wn + nt * 16 + m15) * 40 + khi * 16;
      bF[nt].h[0] = *(const bf16x8*)p;
      bF[nt].h[1] = *(const bf16x8*)(p + 8);
    }
    for (int mt = 0; mt < 4; ++mt)
      for (int nt = 0; nt < 2; ++nt)
        acc[mt][nt] = wmma_bf16(aF[mt], bF[nt], acc[mt][nt]);
  }
#else
  // Fallback: synchronous staging through VGPRs (round-1 path)
  const int lrow = tid >> 1;         // staging row 0..127
  const int lcol = (tid & 1) * 16;   // staging col chunk 0/16
  for (int kk = 0; kk < K; kk += 32) {
    const __bf16* ag = A  + (size_t)(mwg + lrow) * lda + kk + lcol;
    const __bf16* bg = Bt + (size_t)(nwg + lrow) * K   + kk + lcol;
    bf16x8 a0 = *(const bf16x8*)ag;
    bf16x8 a1 = *(const bf16x8*)(ag + 8);
    bf16x8 b0 = *(const bf16x8*)bg;
    bf16x8 b1 = *(const bf16x8*)(bg + 8);
    __syncthreads();
    *(bf16x8*)(As[0] + lrow * 40 + lcol)     = a0;
    *(bf16x8*)(As[0] + lrow * 40 + lcol + 8) = a1;
    *(bf16x8*)(Bs[0] + lrow * 40 + lcol)     = b0;
    *(bf16x8*)(Bs[0] + lrow * 40 + lcol + 8) = b1;
    __syncthreads();
    Frag16 aF[4], bF[2];
    for (int mt = 0; mt < 4; ++mt) {
      const __bf16* p = As[0] + (wm + mt * 16 + m15) * 40;
      aF[mt].h[0] = *(const bf16x8*)(p + khi * 16);
      aF[mt].h[1] = *(const bf16x8*)(p + khi * 16 + 8);
    }
    for (int nt = 0; nt < 2; ++nt) {
      const __bf16* p = Bs[0] + (wn + nt * 16 + m15) * 40 + khi * 16;
      bF[nt].h[0] = *(const bf16x8*)p;
      bF[nt].h[1] = *(const bf16x8*)(p + 8);
    }
    for (int mt = 0; mt < 4; ++mt)
      for (int nt = 0; nt < 2; ++nt)
        acc[mt][nt] = wmma_bf16(aF[mt], bF[nt], acc[mt][nt]);
  }
#endif

  for (int mt = 0; mt < 4; ++mt)
    for (int nt = 0; nt < 2; ++nt) {
      int n = nwg + wn + nt * 16 + m15;
      float bb = bias[n];
      for (int r = 0; r < 8; ++r) {
        int m = mwg + wm + mt * 16 + r + khi * 8;
        float vv = acc[mt][nt][r] + bb;
        if (outF32) ((float*)Cout)[(size_t)m * 1024 + n] = vv;
        else        ((__bf16*)Cout)[(size_t)m * 1024 + n] = f2bf(vv);
      }
    }
}

// ---------------------------------------------------------------------------
// 4) Flash attention. grid(64 bh, 8 qblocks), 256 thr = 8 waves x 16 q-rows.
//    qk ws: [8192][1024] (q cols 0..511, k cols 512..1023), v ws: [8192][1024]
// ---------------------------------------------------------------------------
__global__ __launch_bounds__(256) void k_attn(const __bf16* __restrict__ qk,
                                              const __bf16* __restrict__ v,
                                              __bf16* __restrict__ o) {
  __shared__ __attribute__((aligned(16))) __bf16 vT[128 * 40];    // [dim][key]
  __shared__ __attribute__((aligned(16))) __bf16 Ps[8 * 16 * 40]; // per-wave P

  const int tid = threadIdx.x, lane = tid & 31, wave = tid >> 5;
  const int bh = blockIdx.x, b = bh >> 3, h = bh & 7;
  const int qrow0 = b * SEQ + blockIdx.y * 128 + wave * 16;  // global row
  const int m15 = lane & 15, khi = lane >> 4, klo8 = khi * 8;

  // q fragments (K=64 -> 2 A-fragments), loaded once
  Frag16 qA[2];
  {
    const __bf16* qp = qk + (size_t)(qrow0 + m15) * 1024 + h * DIM_HEAD;
    for (int f = 0; f < 2; ++f) {
      qA[f].h[0] = *(const bf16x8*)(qp + f * 32 + khi * 16);
      qA[f].h[1] = *(const bf16x8*)(qp + f * 32 + khi * 16 + 8);
    }
  }

  float mrun[8], lrun[8];
  f32x8 acc[8];
  for (int r = 0; r < 8; ++r) { mrun[r] = -1e30f; lrun[r] = 0.0f; }
  for (int nt = 0; nt < 8; ++nt) acc[nt] = zero8();

  const int vkey = tid >> 3;          // staging: key 0..31
  const int vdc  = (tid & 7) * 16;    // staging: dim chunk

  for (int j0 = 0; j0 < SEQ; j0 += 32) {
    __syncthreads();
    { // stage v block transposed: vT[dim][key]
      const __bf16* vp = v + (size_t)(b * SEQ + j0 + vkey) * 1024 + h * DV + vdc;
      bf16x8 c0 = *(const bf16x8*)vp;
      bf16x8 c1 = *(const bf16x8*)(vp + 8);
      for (int i = 0; i < 8; ++i) vT[(vdc + i) * 40 + vkey]     = c0[i];
      for (int i = 0; i < 8; ++i) vT[(vdc + 8 + i) * 40 + vkey] = c1[i];
    }
    __syncthreads();

    // S = q . k^T for two 16-key sub-blocks (k B-fragments direct from global)
    f32x8 s[2];
    for (int sub = 0; sub < 2; ++sub) {
      const __bf16* kp = qk + (size_t)(b * SEQ + j0 + sub * 16 + m15) * 1024
                       + INNER + h * DIM_HEAD + khi * 16;
      Frag16 kB0, kB1;
      kB0.h[0] = *(const bf16x8*)kp;
      kB0.h[1] = *(const bf16x8*)(kp + 8);
      kB1.h[0] = *(const bf16x8*)(kp + 32);
      kB1.h[1] = *(const bf16x8*)(kp + 40);
      f32x8 t = zero8();
      t = wmma_bf16(qA[0], kB0, t);
      t = wmma_bf16(qA[1], kB1, t);
      s[sub] = t * SCALE;
    }

    // online softmax: row r of this wave lives across the lane-16 group
    for (int r = 0; r < 8; ++r) {
      float t = fmaxf(s[0][r], s[1][r]);
      t = fmaxf(t, __shfl_xor(t, 1, 32));
      t = fmaxf(t, __shfl_xor(t, 2, 32));
      t = fmaxf(t, __shfl_xor(t, 4, 32));
      t = fmaxf(t, __shfl_xor(t, 8, 32));
      float mnew = fmaxf(mrun[r], t);
      float c = __expf(mrun[r] - mnew);
      mrun[r] = mnew;
      lrun[r] *= c;
      for (int nt = 0; nt < 8; ++nt) acc[nt][r] *= c;
      float e0 = __expf(s[0][r] - mnew);
      float e1 = __expf(s[1][r] - mnew);
      float rs = e0 + e1;
      rs += __shfl_xor(rs, 1, 32);
      rs += __shfl_xor(rs, 2, 32);
      rs += __shfl_xor(rs, 4, 32);
      rs += __shfl_xor(rs, 8, 32);
      lrun[r] += rs;
      s[0][r] = e0; s[1][r] = e1;
    }

    // C-layout -> A-fragment reshape of P through per-wave LDS
    __bf16* ps = Ps + wave * 16 * 40;
    for (int sub = 0; sub < 2; ++sub)
      for (int r = 0; r < 8; ++r)
        ps[(r + khi * 8) * 40 + sub * 16 + m15] = f2bf(s[sub][r]);
    __builtin_amdgcn_wave_barrier();
    Frag16 pA;
    pA.h[0] = *(const bf16x8*)(ps + m15 * 40 + khi * 16);
    pA.h[1] = *(const bf16x8*)(ps + m15 * 40 + khi * 16 + 8);

    // acc += P (16x32) x v (32x128): 8 N-tiles
    for (int nt = 0; nt < 8; ++nt) {
      const __bf16* vp = vT + (nt * 16 + m15) * 40 + khi * 16;
      Frag16 vB;
      vB.h[0] = *(const bf16x8*)vp;
      vB.h[1] = *(const bf16x8*)(vp + 8);
      acc[nt] = wmma_bf16(pA, vB, acc[nt]);
    }
  }

  // epilogue: normalize and store concat-head layout [8192][1024]
  for (int r = 0; r < 8; ++r) {
    float inv = 1.0f / lrun[r];
    size_t row = (size_t)(qrow0 + r + khi * 8);
    __bf16* op = o + row * 1024 + h * DV;
    for (int nt = 0; nt < 8; ++nt)
      op[nt * 16 + m15] = f2bf(acc[nt][r] * inv);
  }
}

// ---------------------------------------------------------------------------
extern "C" void kernel_launch(void* const* d_in, const int* in_sizes, int n_in,
                              void* d_out, int out_size, void* d_ws, size_t ws_size,
                              hipStream_t stream) {
  (void)in_sizes; (void)n_in; (void)out_size; (void)ws_size;
  const float* x   = (const float*)d_in[0];
  const float* Wqk = (const float*)d_in[1];
  const float* bqk = (const float*)d_in[2];
  const float* Wv  = (const float*)d_in[3];
  const float* bv  = (const float*)d_in[4];
  const float* Wo  = (const float*)d_in[5];
  const float* bo  = (const float*)d_in[6];

  char* ws = (char*)d_ws;
  size_t off = 0;
  __bf16* xbf  = (__bf16*)(ws + off); off += (size_t)ROWS * XW * 2;       // 16 MiB
  __bf16* wqkt = (__bf16*)(ws + off); off += (size_t)1024 * 512 * 2;      //  1 MiB
  __bf16* wvt  = (__bf16*)(ws + off); off += (size_t)1024 * 1024 * 2;     //  2 MiB
  __bf16* wot  = (__bf16*)(ws + off); off += (size_t)1024 * 1024 * 2;     //  2 MiB
  __bf16* qkws = (__bf16*)(ws + off); off += (size_t)ROWS * 1024 * 2;     // 16 MiB
  __bf16* vws  = (__bf16*)(ws + off); off += (size_t)ROWS * 1024 * 2;     // 16 MiB
  __bf16* ows  = (__bf16*)(ws + off); off += (size_t)ROWS * 1024 * 2;     // 16 MiB

  // prep: x -> bf16, weights -> transposed bf16
  k_cvt_bf16<<<(ROWS * XW) / (256 * 8), 256, 0, stream>>>(x, xbf);
  k_transpose_bf16<<<dim3(1024 / 32, 512 / 32),  256, 0, stream>>>(Wqk, wqkt, 512, 1024);
  k_transpose_bf16<<<dim3(1024 / 32, 1024 / 32), 256, 0, stream>>>(Wv,  wvt,  1024, 1024);
  k_transpose_bf16<<<dim3(1024 / 32, 1024 / 32), 256, 0, stream>>>(Wo,  wot,  1024, 1024);

  // qk projection: A = last 512 cols of x (pointer offset, lda=1024), K=512
  k_gemm_bf16<<<dim3(8, 64), 256, 0, stream>>>(xbf + CLS, XW, wqkt, bqk, qkws, 0, 512);
  // v projection: K=1024
  k_gemm_bf16<<<dim3(8, 64), 256, 0, stream>>>(xbf, XW, wvt, bv, vws, 0, 1024);
  // flash attention over 64 (b,h) pairs, 8 query blocks each
  k_attn<<<dim3(64, 8), 256, 0, stream>>>(qkws, vws, ows);
  // output projection: fp32 epilogue straight to d_out
  k_gemm_bf16<<<dim3(8, 64), 256, 0, stream>>>(ows, 1024, wot, bo, d_out, 1, 1024);
}
